// GraphAttentionLayer_hyper_25503515803781
// MI455X (gfx1250) — matrix operand order repaired and checked by reading, not compiled, not run on hardware
//
#include <hip/hip_runtime.h>

#define Bb    8
#define Nn_   2048
#define HYPER 128
#define Ff    64
#define ALPHA 0.2f
#define NEGINF (-9.0e15f)

typedef __attribute__((ext_vector_type(16))) _Float16 v16h;
typedef __attribute__((ext_vector_type(8)))  _Float16 v8h;
typedef __attribute__((ext_vector_type(8)))  float    v8f;
typedef unsigned __attribute__((ext_vector_type(4)))  u32x4;
typedef int      __attribute__((ext_vector_type(4)))  i32x4;
typedef int      __attribute__((ext_vector_type(8)))  i32x8;

#if __has_builtin(__builtin_amdgcn_tensor_load_to_lds)
#define HAS_TDM 1
#if __has_include(<hip/amd_detail/amd_gfx1250_TDM.h>)
#define TDM_SIX_ARG 1   // therock-10 headers => 6-arg builtin
#else
#define TDM_SIX_ARG 0   // ROCm 7.2 => 5-arg builtin
#endif
#else
#define HAS_TDM 0
#endif

// Pack two contiguous 8 x f16 LDS runs into a WMMA v16h operand.
// Lets the compiler emit ds_load_b128 x2 instead of 16x ds_load_u16.
__device__ inline v16h pack_ab(const _Float16* lo8, const _Float16* hi8) {
  v8h lo = *(const v8h*)lo8;
  v8h hi = *(const v8h*)hi8;
  v16h r;
#pragma unroll
  for (int e = 0; e < 8; ++e) { r[e] = lo[e]; r[e + 8] = hi[e]; }
  return r;
}

// ---- Kernel A: hypernetwork weights W[b,i,o] = |state@Wp_w^T + b|  (f16) ----
__global__ void k_hyper(const float* __restrict__ state, const float* __restrict__ Wp_w,
                        const float* __restrict__ Wp_b, _Float16* __restrict__ Wf16) {
  int b = blockIdx.x;
  __shared__ float ss[HYPER];
  for (int i = threadIdx.x; i < HYPER; i += 256) ss[i] = state[b * HYPER + i];
  __syncthreads();
  for (int io = threadIdx.x; io < Ff * Ff; io += 256) {
    const float* wr = Wp_w + (size_t)io * HYPER;
    float acc = Wp_b[io];
    for (int k = 0; k < HYPER; ++k) acc += ss[k] * wr[k];
    Wf16[b * Ff * Ff + io] = (_Float16)fabsf(acc);
  }
}

// ---- Kernel Pack: adj (134MB, read ONCE) -> 4MB bitmask, ballot over n ----
__global__ void k_pack(const int* __restrict__ adj, unsigned* __restrict__ packed) {
  int g = blockIdx.x * blockDim.x + threadIdx.x;
  int lane = g & 31;
  int w = g >> 5;                      // word index = (b*N+m)*64 + nw
  size_t row = (size_t)(w >> 6);       // b*N + m
  int n = ((w & 63) << 5) + lane;
  int v = adj[row * Nn_ + n];
  unsigned m = __builtin_amdgcn_ballot_w32(v > 0);
  if (lane == 0) packed[w] = m;
}

// ---- Kernel B: Wh = h @ W via v_wmma_f32_16x16x32_f16 ----
__global__ void k_wh(const float* __restrict__ h, const _Float16* __restrict__ Wf16,
                     float* __restrict__ Wh) {
  int b = blockIdx.y;
  int r0 = blockIdx.x * 16;
  int t = threadIdx.x, lane = t & 31, wv = t >> 5;   // 128 thr = 4 waves
  __shared__ __align__(16) _Float16 hs[16][Ff];      // [row][i]
  __shared__ __align__(16) _Float16 WsT[Ff][Ff];     // transposed: [o][i]
  for (int i = t; i < 16 * Ff; i += 128) {
    int rr = i >> 6, cc = i & 63;
    hs[rr][cc] = (_Float16)h[((size_t)b * Nn_ + r0 + rr) * Ff + cc];
  }
  for (int i = t; i < Ff * Ff; i += 128) {
    int in = i >> 6, o = i & 63;
    WsT[o][in] = Wf16[b * Ff * Ff + i];
  }
  __syncthreads();
  int Mrow = lane & 15, hh = lane >> 4, col = wv * 16 + (lane & 15);
  v8f acc = {};
#pragma unroll
  for (int kk = 0; kk < 2; ++kk) {
    // A 16x32: lane's elems = two contiguous runs of 8
    const _Float16* ar = &hs[Mrow][kk * 32];
    v16h A = pack_ab(ar + hh * 8, ar + 16 + hh * 8);
    // B 32x16 from transposed panel: two contiguous runs of 8 (same K order as A)
    const _Float16* br = &WsT[col][kk * 32];
    v16h Bv = pack_ab(br + hh * 8, br + 16 + hh * 8);
    acc = __builtin_amdgcn_wmma_f32_16x16x32_f16(false, A, false, Bv, (short)0, acc, false, false);
  }
#pragma unroll
  for (int r = 0; r < 8; ++r) {
    int M = r + hh * 8;    // C layout: VGPR r, lanes>=16 -> M+8
    Wh[((size_t)b * Nn_ + r0 + M) * Ff + wv * 16 + (lane & 15)] = acc[r];
  }
}

// ---- Kernel C: Wh1/Wh2 = Wh @ a-halves ----
__global__ void k_attvec(const float* __restrict__ Wh, const float* __restrict__ a,
                         float* __restrict__ Wh1, float* __restrict__ Wh2) {
  __shared__ float a1[Ff], a2[Ff];
  int t = threadIdx.x;
  if (t < Ff) a1[t] = a[t];
  else if (t < 2 * Ff) a2[t - Ff] = a[t];
  __syncthreads();
  int idx = blockIdx.x * 256 + t;
  if (idx < Bb * Nn_) {
    const float* row = Wh + (size_t)idx * Ff;
    float s1 = 0.f, s2 = 0.f;
    for (int o = 0; o < Ff; ++o) { float v = row[o]; s1 += v * a1[o]; s2 += v * a2[o]; }
    Wh1[idx] = s1; Wh2[idx] = s2;
  }
}

// ---- Kernel E1: per-column (axis=1 softmax) online max/sum over m ----
__global__ void k_colstats(const float* __restrict__ Wh1, const float* __restrict__ Wh2,
                           const unsigned* __restrict__ packed,
                           float* __restrict__ colmax, float* __restrict__ colsum) {
  int b = blockIdx.y, nw = blockIdx.x;            // 64 n-words of 32 cols
  int t = threadIdx.x, c = t & 31, sl = t >> 5;   // 256 thr: 8 m-slices x 32 cols
  __shared__ float wh1s[Nn_];
  __shared__ unsigned mw[Nn_];
  for (int i = t; i < Nn_; i += 256) {
    wh1s[i] = Wh1[b * Nn_ + i];
    mw[i] = packed[((size_t)b * Nn_ + i) * 64 + nw];
  }
  __syncthreads();
  float w2 = Wh2[b * Nn_ + nw * 32 + c];
  float mx = -3.0e38f, sm = 0.f;
  for (int m = sl * 256; m < sl * 256 + 256; ++m) {
    float pre = wh1s[m] + w2;
    float lr = pre > 0.f ? pre : ALPHA * pre;
    float s = ((mw[m] >> c) & 1u) ? lr : NEGINF;   // finite -9e15 like reference
    if (s > mx) { sm = sm * __expf(mx - s) + 1.f; mx = s; }
    else sm += __expf(s - mx);
  }
  __shared__ float pm[8][32], ps[8][32];
  pm[sl][c] = mx; ps[sl][c] = sm;
  __syncthreads();
  if (t < 32) {
    float M = pm[0][t], S = ps[0][t];
    for (int j = 1; j < 8; ++j) {
      float m2 = pm[j][t], s2 = ps[j][t];
      float nm = fmaxf(M, m2);
      S = S * __expf(M - nm) + s2 * __expf(m2 - nm);
      M = nm;
    }
    colmax[b * Nn_ + nw * 32 + t] = M;
    colsum[b * Nn_ + nw * 32 + t] = S;
  }
}

// ---- Kernel E1b: fold 1/colsum[n] in, store TRANSPOSED f16: WhsT[b][o][n] ----
__global__ void k_scale(const float* __restrict__ Wh, const float* __restrict__ colsum,
                        _Float16* __restrict__ WhsT) {
  int idx = blockIdx.x * 256 + threadIdx.x;   // n fastest for coalesced writes
  int n = idx & (Nn_ - 1);
  int o = (idx >> 11) & (Ff - 1);
  int b = idx >> 17;
  float v = Wh[((size_t)b * Nn_ + n) * Ff + o] / colsum[b * Nn_ + n];
  WhsT[((size_t)b * Ff + o) * Nn_ + n] = (_Float16)v;
}

// ---- Kernel E2: h_prime[m,o] = sum_n exp(s[m,n]-colmax[n]) * Wh'[n,o], ELU ----
__global__ void k_gat(const float* __restrict__ Wh1, const float* __restrict__ Wh2,
                      const float* __restrict__ colmax, const unsigned* __restrict__ packed,
                      const _Float16* __restrict__ WhsT, float* __restrict__ out) {
  int b = blockIdx.y;
  int m0 = blockIdx.x * 16;
  int t = threadIdx.x, lane = t & 31, wv = t >> 5;   // 128 thr = 4 waves, wave=o-tile
  __shared__ float wh2s[Nn_];
  __shared__ float cmax[Nn_];
  __shared__ unsigned mrow[16][64];
  __shared__ __align__(16) _Float16 BshT[Ff][32];    // [o][n-local], TDM tile target
  __shared__ float wh1s[16];
  for (int i = t; i < Nn_; i += 128) {
    wh2s[i] = Wh2[b * Nn_ + i];
    cmax[i] = colmax[b * Nn_ + i];
  }
  for (int i = t; i < 16 * 64; i += 128) {
    int r = i >> 6, w = i & 63;
    mrow[r][w] = packed[((size_t)b * Nn_ + m0 + r) * 64 + w];
  }
  if (t < 16) wh1s[t] = Wh1[b * Nn_ + m0 + t];
  __syncthreads();

  int Mrow = lane & 15, hh = lane >> 4, col = wv * 16 + (lane & 15);
  float myWh1 = wh1s[Mrow];
  v8f acc = {};
  for (int n0 = 0; n0 < Nn_; n0 += 32) {
    __syncthreads();   // all waves done reading previous B panel
#if HAS_TDM
    if (wv == 0) {
      // Tensor Data Mover: 2-D tile (32 n) x (64 o), f16, row stride 2048,
      // linear LDS fill == BshT[64][32].  D# per CDNA5 ISA.
      unsigned ldsoff = (unsigned)(size_t)(void*)&BshT[0][0];
      unsigned long long ga =
          (unsigned long long)(size_t)(const void*)(WhsT + (size_t)b * Ff * Nn_ + n0);
      u32x4 g0;
      g0[0] = 1u;                                            // count=1, user mode
      g0[1] = ldsoff;                                        // lds_addr
      g0[2] = (unsigned)(ga & 0xffffffffu);                  // global_addr[31:0]
      g0[3] = (unsigned)((ga >> 32) & 0x1ffffffu) | (2u << 30); // ga[56:32] | type=2
      i32x8 g1;
      g1[0] = (int)(1u << 16);                               // data_size=1 (2 bytes)
      g1[1] = (int)(((unsigned)Nn_ & 0xffffu) << 16);        // tensor_dim0 lo16
      g1[2] = (int)(((unsigned)Ff & 0xffffu) << 16);         // dim0 hi=0 | tensor_dim1 lo16
      g1[3] = (int)(32u << 16);                              // dim1 hi=0 | tile_dim0=32
      g1[4] = (int)Ff;                                       // tile_dim1=64, tile_dim2=0
      g1[5] = (int)Nn_;                                      // tensor_dim0_stride lo32
      g1[6] = 0;
      g1[7] = 0;
      i32x4 z4 = {};
#if TDM_SIX_ARG
      i32x8 z8 = {};
      __builtin_amdgcn_tensor_load_to_lds(g0, g1, z4, z4, z8, 0);
#else
      __builtin_amdgcn_tensor_load_to_lds(g0, g1, z4, z4, 0);
#endif
      __builtin_amdgcn_s_wait_tensorcnt(0);
    }
#else
    {
      const unsigned* src = (const unsigned*)(WhsT + (size_t)b * Ff * Nn_ + n0);
      unsigned* dst = (unsigned*)&BshT[0][0];
      for (int i = t; i < Ff * 16; i += 128) {     // 64 rows x 16 dwords
        int r = i >> 4, cdw = i & 15;
        dst[r * 16 + cdw] = src[(size_t)r * (Nn_ / 2) + cdw];
      }
    }
#endif
    __syncthreads();
    if (n0 + 32 < Nn_)
      __builtin_prefetch((const void*)(WhsT + (size_t)b * Ff * Nn_ + (t & 63) * Nn_ + n0 + 32), 0, 1);

    // compute A tile (16m x 32n) in registers, WMMA f16 A layout:
    // elem e<8 -> K = hh*8+e ; e>=8 -> K = 16 + hh*8 + (e-8)
    v16h A;
#pragma unroll
    for (int e = 0; e < 16; ++e) {
      int k = (e < 8) ? (hh * 8 + e) : (16 + hh * 8 + (e - 8));
      int n = n0 + k;
      float pre = myWh1 + wh2s[n];
      float lr = pre > 0.f ? pre : ALPHA * pre;
      unsigned bit = (mrow[Mrow][n >> 5] >> (n & 31)) & 1u;
      float s = bit ? lr : NEGINF;
      A[e] = (_Float16)__expf(s - cmax[n]);
    }
    const _Float16* br = &BshT[col][0];
    v16h Bv = pack_ab(br + hh * 8, br + 16 + hh * 8);
    acc = __builtin_amdgcn_wmma_f32_16x16x32_f16(false, A, false, Bv, (short)0, acc, false, false);
  }
#pragma unroll
  for (int r = 0; r < 8; ++r) {
    int M = r + hh * 8;
    float x = acc[r];
    float y = x > 0.f ? x : (__expf(x) - 1.f);   // ELU
    out[((size_t)b * Nn_ + m0 + M) * Ff + wv * 16 + (lane & 15)] = y;
  }
}

extern "C" void kernel_launch(void* const* d_in, const int* in_sizes, int n_in,
                              void* d_out, int out_size, void* d_ws, size_t ws_size,
                              hipStream_t stream) {
  const float* state = (const float*)d_in[0];
  const float* h     = (const float*)d_in[1];
  const int*   adj   = (const int*)d_in[2];
  const float* Wp_w  = (const float*)d_in[3];
  const float* Wp_b  = (const float*)d_in[4];
  const float* a     = (const float*)d_in[5];
  float* out = (float*)d_out;

  char* ws = (char*)d_ws;
  float*    Wh     = (float*)(ws);                         // 4,194,304 B
  _Float16* WhsT   = (_Float16*)(ws + 4194304);            // 2,097,152 B
  _Float16* Wf16   = (_Float16*)(ws + 6291456);            //    65,536 B
  float*    Wh1    = (float*)(ws + 6356992);               //    65,536 B
  float*    Wh2    = (float*)(ws + 6422528);               //    65,536 B
  float*    cmaxp  = (float*)(ws + 6488064);               //    65,536 B
  float*    csump  = (float*)(ws + 6553600);               //    65,536 B
  unsigned* packed = (unsigned*)(ws + 6619136);            // 4,194,304 B

  k_hyper<<<Bb, 256, 0, stream>>>(state, Wp_w, Wp_b, Wf16);
  k_pack<<<(Bb * Nn_ * 64 * 32) / 256, 256, 0, stream>>>(adj, packed);
  k_wh<<<dim3(Nn_ / 16, Bb), 128, 0, stream>>>(h, Wf16, Wh);
  k_attvec<<<(Bb * Nn_) / 256, 256, 0, stream>>>(Wh, a, Wh1, Wh2);
  k_colstats<<<dim3(64, Bb), 256, 0, stream>>>(Wh1, Wh2, packed, cmaxp, csump);
  k_scale<<<(Bb * Nn_ * Ff) / 256, 256, 0, stream>>>(Wh, csump, WhsT);
  k_gat<<<dim3(Nn_ / 16, Bb), 128, 0, stream>>>(Wh1, Wh2, cmaxp, packed, WhsT, out);
}
